// NonLocalSelfAttention_24232205484147
// MI455X (gfx1250) — matrix operand order repaired and checked by reading, not compiled, not run on hardware
//
#include <hip/hip_runtime.h>

// ---------------------------------------------------------------------------
// NonLocalSelfAttention for MI455X (gfx1250, wave32, WMMA bf16, async-to-LDS)
// B=16, C=256, H=W=32 (HW=1024), heads=8, head_dim=64, inner=512, TD=1024
// ---------------------------------------------------------------------------

#define B_SZ 16
#define C_SZ 256
#define HW 1024
#define HEADS 8
#define HDIM 64
#define INNER 512
#define TD 1024
#define GROUPS 32

typedef __attribute__((ext_vector_type(16))) __bf16 v16bf;
typedef __attribute__((ext_vector_type(8)))  float  v8f;

union FragU {
    v16bf v;
    uint4 q[2];
};

// D = A*B + C, bf16 inputs, f32 accum (emits v_wmma_f32_16x16x32_bf16)
__device__ __forceinline__ v8f wmma_bf16(v16bf a, v16bf b, v8f c) {
    return __builtin_amdgcn_wmma_f32_16x16x32_bf16(
        /*neg_a=*/false, a, /*neg_b=*/false, b,
        /*c_mod=*/(short)0, c, /*reuse_a=*/false, /*reuse_b=*/false);
}

// A fragment (16x32 bf16, row-major source, row stride lda elements).
// ISA layout: lane l: M = l&15; kb = (l<16)?0:8; elems j<8 -> K=kb+j,
// j>=8 -> K=kb+16+(j-8).  => two contiguous 16B loads per lane.
__device__ __forceinline__ v16bf load_fragA(const __bf16* base, int lda, int lane) {
    const int m  = lane & 15;
    const int kb = (lane < 16) ? 0 : 8;
    const __bf16* p = base + (size_t)m * lda + kb;
    FragU f;
    f.q[0] = *reinterpret_cast<const uint4*>(p);
    f.q[1] = *reinterpret_cast<const uint4*>(p + 16);
    return f.v;
}

// B fragment (32x16 bf16) from a K-contiguous layout: source row n holds the
// 32 K-values contiguously (row stride ldb elements).
// ISA layout: lane l: N = l&15; kb = (l<16)?0:8; elems j<8 -> K=kb+j,
// j>=8 -> K=kb+16+(j-8).
__device__ __forceinline__ v16bf load_fragB(const __bf16* base, int ldb, int lane) {
    const int n  = lane & 15;
    const int kb = (lane < 16) ? 0 : 8;
    const __bf16* p = base + (size_t)n * ldb + kb;
    FragU f;
    f.q[0] = *reinterpret_cast<const uint4*>(p);
    f.q[1] = *reinterpret_cast<const uint4*>(p + 16);
    return f.v;
}

// ---------------------------------------------------------------------------
// Kernel 0: f32 -> bf16 weight conversion
// ---------------------------------------------------------------------------
__global__ void convert_bf16_kernel(const float* __restrict__ src,
                                    __bf16* __restrict__ dst, int n) {
    int i = blockIdx.x * blockDim.x + threadIdx.x;
    if (i < n) dst[i] = (__bf16)src[i];
}

// ---------------------------------------------------------------------------
// Kernel 1: emb[b,oc] = embedding[b,:] . emb_w[oc,:] + emb_b[oc]
// ---------------------------------------------------------------------------
__global__ void emb_kernel(const float* __restrict__ embedding,
                           const float* __restrict__ ew,
                           const float* __restrict__ eb,
                           float* __restrict__ emb) {
    const int b  = blockIdx.x;
    const int oc = threadIdx.x;
    const float* e  = embedding + (size_t)b * TD;
    const float* w2 = ew + (size_t)oc * TD;
    float acc = eb[oc];
    for (int t = 0; t < TD; t += 4) {
        acc += e[t + 0] * w2[t + 0];
        acc += e[t + 1] * w2[t + 1];
        acc += e[t + 2] * w2[t + 2];
        acc += e[t + 3] * w2[t + 3];
    }
    emb[(size_t)b * C_SZ + oc] = acc;
}

// ---------------------------------------------------------------------------
// Kernel 2: GroupNorm -> xn transposed [B, HW, C] in bf16
// ---------------------------------------------------------------------------
__global__ void groupnorm_kernel(const float* __restrict__ x,
                                 const float* __restrict__ nw,
                                 const float* __restrict__ nb,
                                 __bf16* __restrict__ xnt) {
    const int g = blockIdx.x, b = blockIdx.y;
    const int c0 = g * (C_SZ / GROUPS);               // 8 channels per group
    const float* xb = x + ((size_t)b * C_SZ + c0) * HW;

    float s = 0.f, ss = 0.f;
    for (int i = threadIdx.x; i < 8 * HW; i += 256) {
        float v = xb[i];
        s += v; ss += v * v;
    }
#pragma unroll
    for (int off = 16; off > 0; off >>= 1) {
        s  += __shfl_xor(s,  off, 32);
        ss += __shfl_xor(ss, off, 32);
    }
    __shared__ float red[16];
    const int wave = threadIdx.x >> 5, lane = threadIdx.x & 31;
    if (lane == 0) { red[wave] = s; red[8 + wave] = ss; }
    __syncthreads();
    if (threadIdx.x == 0) {
        float ts = 0.f, tss = 0.f;
#pragma unroll
        for (int i = 0; i < 8; ++i) { ts += red[i]; tss += red[8 + i]; }
        const float mean = ts * (1.f / 8192.f);
        const float var  = tss * (1.f / 8192.f) - mean * mean;
        red[0] = mean;
        red[1] = rsqrtf(var + 1e-5f);
    }
    __syncthreads();
    const float mean = red[0], rstd = red[1];

    for (int i = threadIdx.x; i < 8 * HW; i += 256) {
        const int c = i & 7;
        const int p = i >> 3;
        const float v = xb[(size_t)c * HW + p];
        const float y = (v - mean) * rstd * nw[c0 + c] + nb[c0 + c];
        xnt[((size_t)b * HW + p) * C_SZ + c0 + c] = (__bf16)y;
    }
}

// ---------------------------------------------------------------------------
// Kernel 3: QKV GEMM  qkv[o,p] = in_w[o,:] . xn[:,p] + in_b[o]
// Q,K stored transposed [B,H,HW,64]; V natural [B,H,64,HW].
// Q is pre-scaled by 1/sqrt(head_dim) so attention skips the scale.
// ---------------------------------------------------------------------------
__global__ void qkv_kernel(const __bf16* __restrict__ wbf,   // [1536,256]
                           const float*  __restrict__ bias,  // [1536]
                           const __bf16* __restrict__ xnt,   // [B,HW,C]
                           __bf16* __restrict__ qt,
                           __bf16* __restrict__ kt,
                           __bf16* __restrict__ vbuf) {
    const int b    = blockIdx.z;
    const int o0   = blockIdx.y * 128 + (threadIdx.x >> 5) * 16;
    const int p0   = blockIdx.x * 32;
    const int lane = threadIdx.x & 31;
    const __bf16* xb = xnt + ((size_t)b * HW + p0) * C_SZ;

    v8f acc0 = {}, acc1 = {};
    for (int kk = 0; kk < C_SZ; kk += 32) {
        v16bf a  = load_fragA(wbf + (size_t)o0 * C_SZ + kk, C_SZ, lane);
        v16bf b0 = load_fragB(xb + kk, C_SZ, lane);
        v16bf b1 = load_fragB(xb + (size_t)16 * C_SZ + kk, C_SZ, lane);
        acc0 = wmma_bf16(a, b0, acc0);
        acc1 = wmma_bf16(a, b1, acc1);
    }

    const int mb = (lane < 16) ? 0 : 8;
    const int nl = lane & 15;
#pragma unroll
    for (int f = 0; f < 2; ++f) {
        const v8f acc = f ? acc1 : acc0;
        const int p = p0 + f * 16 + nl;
#pragma unroll
        for (int r = 0; r < 8; ++r) {
            const int o = o0 + mb + r;
            const float val = acc[r] + bias[o];
            const int sec = o >> 9;       // 0=q 1=k 2=v
            const int rem = o & 511;
            const int h = rem >> 6;
            const int c = rem & 63;
            const size_t bh = (size_t)b * HEADS + h;
            if (sec == 0)
                qt[(bh * HW + p) * HDIM + c] = (__bf16)(val * 0.125f);
            else if (sec == 1)
                kt[(bh * HW + p) * HDIM + c] = (__bf16)val;
            else
                vbuf[(bh * HDIM + c) * HW + p] = (__bf16)val;
        }
    }
}

// ---------------------------------------------------------------------------
// Kernel 4: flash-style attention per (b,h). Block = 8 waves x 16 rows = 128
// query rows; 64 key columns per step; K/V staged in LDS via CDNA5
// global_load_async_to_lds_b128 with double buffering (ASYNCcnt).
// ---------------------------------------------------------------------------
__device__ __forceinline__ void async_copy_kv(const __bf16* k_n0,   // kt_bh + n0*64
                                              const __bf16* v_n0,   // vbuf_bh + n0
                                              __bf16 (*lk)[64],
                                              __bf16 (*lv)[64], int tid) {
    const unsigned long long ks = (unsigned long long)k_n0;
    const unsigned long long vs = (unsigned long long)v_n0;
#pragma unroll
    for (int s2 = 0; s2 < 2; ++s2) {
        const int q   = tid + s2 * 256;      // 512 16B chunks per 8KB tile
        const int row = q >> 3;
        const int ce  = (q & 7) * 8;         // element offset in row
        // K tile: rows of kt (stride 64 elems) -> lds_k[row][ce]
        const unsigned lka = (unsigned)(unsigned long long)&lk[row][ce];
        const unsigned kgo = (unsigned)((row * HDIM + ce) * 2);
        asm volatile("global_load_async_to_lds_b128 %0, %1, %2"
                     :: "v"(lka), "v"(kgo), "s"(ks) : "memory");
        // V tile: rows of vbuf (stride 1024 elems) -> lds_v[row][ce]
        const unsigned lva = (unsigned)(unsigned long long)&lv[row][ce];
        const unsigned vgo = (unsigned)((row * HW + ce) * 2);
        asm volatile("global_load_async_to_lds_b128 %0, %1, %2"
                     :: "v"(lva), "v"(vgo), "s"(vs) : "memory");
    }
}

__global__ void attn_kernel(const __bf16* __restrict__ qt,
                            const __bf16* __restrict__ kt,
                            const __bf16* __restrict__ vbuf,
                            __bf16* __restrict__ ot) {
    const int b = blockIdx.z, h = blockIdx.y;
    const size_t bh = (size_t)b * HEADS + h;
    const int w    = threadIdx.x >> 5;
    const int lane = threadIdx.x & 31;
    const int tid  = threadIdx.x;
    const int m0   = blockIdx.x * 128 + w * 16;

    __shared__ __bf16 lds_k[2][64][64];   // [buf][n][c]   8KB each
    __shared__ __bf16 lds_v[2][64][64];   // [buf][c][n]   8KB each
    __shared__ __bf16 lds_p[8][16][64];   // per-wave P tile (D->A relayout)

    const __bf16* kt_bh = kt + bh * HW * HDIM;
    const __bf16* v_bh  = vbuf + bh * HDIM * HW;

    const __bf16* qbase = qt + (bh * HW + m0) * HDIM;
    const v16bf qa0 = load_fragA(qbase,      HDIM, lane);   // K = 0..31
    const v16bf qa1 = load_fragA(qbase + 32, HDIM, lane);   // K = 32..63

    float rowmax[8], rowsum[8];
#pragma unroll
    for (int r = 0; r < 8; ++r) { rowmax[r] = -3.0e38f; rowsum[r] = 0.f; }
    v8f oacc[4] = {};

    const int mb = (lane < 16) ? 0 : 8;
    const int nl = lane & 15;

    // prefetch first K/V tile
    async_copy_kv(kt_bh, v_bh, lds_k[0], lds_v[0], tid);

    for (int it = 0; it < HW / 64; ++it) {
        // wait for this iteration's tile, make visible to all waves
        asm volatile("s_wait_asynccnt 0x0" ::: "memory");
        __syncthreads();
        if (it + 1 < HW / 64)
            async_copy_kv(kt_bh + (size_t)(it + 1) * 64 * HDIM,
                          v_bh + (size_t)(it + 1) * 64,
                          lds_k[(it + 1) & 1], lds_v[(it + 1) & 1], tid);

        __bf16 (*lk)[64] = lds_k[it & 1];
        __bf16 (*lv)[64] = lds_v[it & 1];

        // ---- S = Q K^T : four 16x16 column tiles, K-dim 64 ----
        v8f sv[4];
#pragma unroll
        for (int f = 0; f < 4; ++f) {
            v8f s = {};
            s = wmma_bf16(qa0, load_fragB(&lk[f * 16][0],  HDIM, lane), s);
            s = wmma_bf16(qa1, load_fragB(&lk[f * 16][32], HDIM, lane), s);
            sv[f] = s;
        }

        // ---- online softmax (rows live in 16-lane halves) ----
        float alpha[8];
#pragma unroll
        for (int r = 0; r < 8; ++r) {
            float t = fmaxf(fmaxf(sv[0][r], sv[1][r]), fmaxf(sv[2][r], sv[3][r]));
#pragma unroll
            for (int off = 1; off < 16; off <<= 1)
                t = fmaxf(t, __shfl_xor(t, off, 16));
            const float nm = fmaxf(rowmax[r], t);
            alpha[r] = __expf(rowmax[r] - nm);
            rowmax[r] = nm;
            float ps = 0.f;
#pragma unroll
            for (int f = 0; f < 4; ++f) {
                const float pv = __expf(sv[f][r] - nm);
                sv[f][r] = pv;
                ps += pv;
            }
#pragma unroll
            for (int off = 1; off < 16; off <<= 1)
                ps += __shfl_xor(ps, off, 16);
            rowsum[r] = rowsum[r] * alpha[r] + ps;
        }
#pragma unroll
        for (int ct = 0; ct < 4; ++ct)
#pragma unroll
            for (int r = 0; r < 8; ++r)
                oacc[ct][r] *= alpha[r];

        // ---- P: D-layout -> A-layout via per-wave LDS ----
#pragma unroll
        for (int f = 0; f < 4; ++f)
#pragma unroll
            for (int r = 0; r < 8; ++r)
                lds_p[w][mb + r][f * 16 + nl] = (__bf16)sv[f][r];

        v16bf pa0, pa1;
        {
            const int kb2 = (lane < 16) ? 0 : 8;
            const __bf16* p = &lds_p[w][nl][0] + kb2;
            FragU f0, f1;
            f0.q[0] = *reinterpret_cast<const uint4*>(p);
            f0.q[1] = *reinterpret_cast<const uint4*>(p + 16);
            f1.q[0] = *reinterpret_cast<const uint4*>(p + 32);
            f1.q[1] = *reinterpret_cast<const uint4*>(p + 48);
            pa0 = f0.v;
            pa1 = f1.v;
        }

        // ---- O += P V^T : V tile [c][n], K = n contiguous in LDS ----
#pragma unroll
        for (int ct = 0; ct < 4; ++ct) {
            oacc[ct] = wmma_bf16(pa0, load_fragB(&lv[ct * 16][0],  64, lane), oacc[ct]);
            oacc[ct] = wmma_bf16(pa1, load_fragB(&lv[ct * 16][32], 64, lane), oacc[ct]);
        }

        __syncthreads();   // all waves done reading this buffer
    }

    // ---- normalize + store transposed o_t[b][m][h*64 + c] ----
    __bf16* ob = ot + ((size_t)b * HW + m0) * INNER + h * HDIM;
#pragma unroll
    for (int r = 0; r < 8; ++r) {
        const float inv = 1.f / rowsum[r];
#pragma unroll
        for (int ct = 0; ct < 4; ++ct)
            ob[(size_t)(mb + r) * INNER + ct * 16 + nl] = (__bf16)(oacc[ct][r] * inv);
    }
}

// ---------------------------------------------------------------------------
// Kernel 5: out = (out_w @ o + out_b) * emb + x   (fused epilogue)
// ---------------------------------------------------------------------------
__global__ void outproj_kernel(const __bf16* __restrict__ wbf,   // [256,512]
                               const float*  __restrict__ bias,  // [256]
                               const __bf16* __restrict__ ot,    // [B,HW,512]
                               const float*  __restrict__ emb,   // [B,256]
                               const float*  __restrict__ x,     // [B,256,HW]
                               float* __restrict__ out) {
    const int b    = blockIdx.z;
    const int o0   = blockIdx.y * 128 + (threadIdx.x >> 5) * 16;
    const int p0   = blockIdx.x * 32;
    const int lane = threadIdx.x & 31;
    const __bf16* ob = ot + ((size_t)b * HW + p0) * INNER;

    v8f acc0 = {}, acc1 = {};
    for (int kk = 0; kk < INNER; kk += 32) {
        v16bf a  = load_fragA(wbf + (size_t)o0 * INNER + kk, INNER, lane);
        v16bf b0 = load_fragB(ob + kk, INNER, lane);
        v16bf b1 = load_fragB(ob + (size_t)16 * INNER + kk, INNER, lane);
        acc0 = wmma_bf16(a, b0, acc0);
        acc1 = wmma_bf16(a, b1, acc1);
    }

    const int mb = (lane < 16) ? 0 : 8;
    const int nl = lane & 15;
#pragma unroll
    for (int f = 0; f < 2; ++f) {
        const v8f acc = f ? acc1 : acc0;
        const int p = p0 + f * 16 + nl;
#pragma unroll
        for (int r = 0; r < 8; ++r) {
            const int oc = o0 + mb + r;
            const size_t idx = ((size_t)b * C_SZ + oc) * HW + p;
            const float val = acc[r] + bias[oc];
            out[idx] = val * emb[(size_t)b * C_SZ + oc] + x[idx];
        }
    }
}

// ---------------------------------------------------------------------------
// Launch
// ---------------------------------------------------------------------------
extern "C" void kernel_launch(void* const* d_in, const int* in_sizes, int n_in,
                              void* d_out, int out_size, void* d_ws, size_t ws_size,
                              hipStream_t stream) {
    const float* x         = (const float*)d_in[0];
    const float* embedding = (const float*)d_in[1];
    const float* norm_w    = (const float*)d_in[2];
    const float* norm_b    = (const float*)d_in[3];
    const float* in_w      = (const float*)d_in[4];
    const float* in_b      = (const float*)d_in[5];
    const float* out_w     = (const float*)d_in[6];
    const float* out_b     = (const float*)d_in[7];
    const float* emb_w     = (const float*)d_in[8];
    const float* emb_b     = (const float*)d_in[9];
    float* out = (float*)d_out;

    // workspace layout (bytes)
    char* ws = (char*)d_ws;
    __bf16* in_w_bf  = (__bf16*)(ws);                       // 768 KB
    __bf16* out_w_bf = (__bf16*)(ws + (786432));            // 256 KB
    __bf16* xnt      = (__bf16*)(ws + (1048576));           // 8 MB
    __bf16* qt       = (__bf16*)(ws + (9437184));           // 16 MB
    __bf16* kt       = (__bf16*)(ws + (26214400));          // 16 MB
    __bf16* vbuf     = (__bf16*)(ws + (42991616));          // 16 MB
    __bf16* ot       = (__bf16*)(ws + (59768832));          // 16 MB
    float*  embv     = (float*) (ws + (76546048));          // 16 KB

    convert_bf16_kernel<<<(3 * INNER * C_SZ + 255) / 256, 256, 0, stream>>>(
        in_w, in_w_bf, 3 * INNER * C_SZ);
    convert_bf16_kernel<<<(C_SZ * INNER + 255) / 256, 256, 0, stream>>>(
        out_w, out_w_bf, C_SZ * INNER);

    emb_kernel<<<B_SZ, C_SZ, 0, stream>>>(embedding, emb_w, emb_b, embv);

    groupnorm_kernel<<<dim3(GROUPS, B_SZ), 256, 0, stream>>>(x, norm_w, norm_b, xnt);

    qkv_kernel<<<dim3(HW / 32, (3 * INNER) / 128, B_SZ), 256, 0, stream>>>(
        in_w_bf, in_b, xnt, qt, kt, vbuf);

    attn_kernel<<<dim3(HW / 128, HEADS, B_SZ), 256, 0, stream>>>(qt, kt, vbuf, ot);

    outproj_kernel<<<dim3(HW / 32, C_SZ / 128, B_SZ), 256, 0, stream>>>(
        out_w_bf, out_b, ot, embv, x, out);
}
